// SMASelfAttention_15865609191589
// MI455X (gfx1250) — matrix operand order repaired and checked by reading, not compiled, not run on hardware
//
#include <hip/hip_runtime.h>
#include <stdint.h>

// Problem constants (match reference)
#define B_   2
#define S_   2048
#define D_   1024
#define H_   16
#define HD_  64
#define LN_EPS 1e-5f
#define LOG2E 1.4426950408889634f

typedef __attribute__((ext_vector_type(16))) __bf16 v16bf;
typedef __attribute__((ext_vector_type(8)))  float  v8f;

union V16 {
    v16bf v;
    unsigned short us[16];
    uint4 q[2];
};

__device__ __forceinline__ unsigned short f2bf(float f) {
    union { float f; uint32_t u; } x; x.f = f;
    uint32_t u = x.u;
    u += 0x7FFFu + ((u >> 16) & 1u);   // round-to-nearest-even
    return (unsigned short)(u >> 16);
}

// ---------------------------------------------------------------------------
// Kernel 1: LayerNorm (f32 in) -> bf16 rows.  One block per row of D=1024.
// ---------------------------------------------------------------------------
__global__ __launch_bounds__(256)
void ln_bf16_kernel(const float* __restrict__ x,
                    const float* __restrict__ g,
                    const float* __restrict__ be,
                    unsigned short* __restrict__ y)
{
    const int row = blockIdx.x;                 // 0 .. B*S-1
    const int t   = threadIdx.x;
    const float* xr = x + (size_t)row * D_;
    unsigned short* yr = y + (size_t)row * D_;

    float v[4];
    float s = 0.f, s2 = 0.f;
#pragma unroll
    for (int i = 0; i < 4; ++i) {
        v[i] = xr[t + 256 * i];
        s  += v[i];
        s2 += v[i] * v[i];
    }

    __shared__ float red[256];
    red[t] = s; __syncthreads();
    for (int o = 128; o > 0; o >>= 1) { if (t < o) red[t] += red[t + o]; __syncthreads(); }
    const float mean = red[0] * (1.0f / D_);
    __syncthreads();
    red[t] = s2; __syncthreads();
    for (int o = 128; o > 0; o >>= 1) { if (t < o) red[t] += red[t + o]; __syncthreads(); }
    const float var = red[0] * (1.0f / D_) - mean * mean;
    const float inv = rsqrtf(var + LN_EPS);

#pragma unroll
    for (int i = 0; i < 4; ++i) {
        const int c = t + 256 * i;
        yr[c] = f2bf((v[i] - mean) * inv * g[c] + be[c]);
    }
}

// ---------------------------------------------------------------------------
// Kernel 2: weights [in,out] f32 -> out-major bf16  Wt[w][n][k] = W[w][k][n]
// ---------------------------------------------------------------------------
__global__ __launch_bounds__(256)
void wt_bf16_kernel(const float* __restrict__ Wq,
                    const float* __restrict__ Wk,
                    const float* __restrict__ Wv,
                    unsigned short* __restrict__ Wt)
{
    const int idx = blockIdx.x * 256 + threadIdx.x;   // 3 * 1024 * 1024 threads
    const int w   = idx >> 20;
    const int rem = idx & ((1 << 20) - 1);
    const int k   = rem >> 10;
    const int n   = rem & 1023;                       // contiguous -> coalesced read
    const float* W = (w == 0) ? Wq : ((w == 1) ? Wk : Wv);
    Wt[(size_t)w * 1048576 + (size_t)n * 1024 + k] = f2bf(W[(size_t)k * 1024 + n]);
}

// ---------------------------------------------------------------------------
// Kernel 3: QKV projection GEMM with bf16 WMMA.
//   C[4096,1024] = hln[4096,1024] x W[1024,1024] for each of Q,K,V.
//   Block = 4 waves, 64x64 C tile; wave computes 16x64.
//   Q,K stored [B,H,S,HD]; V stored transposed [B,H,HD,S].
//   Q prescaled by (1/sqrt(HD)) * log2(e)  -> softmax runs in exp2 domain.
// ---------------------------------------------------------------------------
__global__ __launch_bounds__(128)
void qkv_wmma_kernel(const unsigned short* __restrict__ hln,  // [4096][1024] bf16
                     const unsigned short* __restrict__ Wt,   // [3][1024][1024] bf16 out-major
                     unsigned short* __restrict__ Qd,
                     unsigned short* __restrict__ Kd,
                     unsigned short* __restrict__ Vt)
{
    const int lane  = threadIdx.x & 31;
    const int wave  = threadIdx.x >> 5;
    const int which = blockIdx.z;                 // 0=Q 1=K 2=V
    const int m0    = blockIdx.x * 64 + wave * 16;
    const int n0    = blockIdx.y * 64;
    const int lh    = lane >> 4;                  // half-wave select
    const int ll    = lane & 15;

    const unsigned short* W = Wt + (size_t)which * (1024 * 1024);

    v8f acc[4];
    const v8f zero = {};
#pragma unroll
    for (int c = 0; c < 4; ++c) acc[c] = zero;

    // A: lane ll holds row m0+ll; halves 0-7 = K[base..base+7], 8-15 = K[base+16..base+23]
    const unsigned short* arow = hln + (size_t)(m0 + ll) * 1024 + lh * 8;

#pragma unroll 2
    for (int k0 = 0; k0 < 1024; k0 += 32) {
        V16 a;
        a.q[0] = *(const uint4*)(arow + k0);
        a.q[1] = *(const uint4*)(arow + k0 + 16);
#pragma unroll
        for (int c = 0; c < 4; ++c) {
            // B: lane holds column n; halves 0-7 = K[base..base+7], 8-15 = K[base+16..base+23]
            const unsigned short* bp = W + (size_t)(n0 + c * 16 + ll) * 1024 + k0 + lh * 8;
            V16 bm;
            bm.q[0] = *(const uint4*)(bp);
            bm.q[1] = *(const uint4*)(bp + 16);
            acc[c] = __builtin_amdgcn_wmma_f32_16x16x32_bf16(
                false, a.v, false, bm.v, (short)0, acc[c], false, false);
        }
    }

    const float scale = (which == 0) ? (0.125f * LOG2E) : 1.0f;
#pragma unroll
    for (int c = 0; c < 4; ++c) {
        const int n  = n0 + c * 16 + ll;
        const int h  = n >> 6;
        const int hd = n & 63;
#pragma unroll
        for (int r = 0; r < 8; ++r) {
            const int m = m0 + r + lh * 8;
            const int b = m >> 11;
            const int s = m & 2047;
            const unsigned short val = f2bf(acc[c][r] * scale);
            if (which == 2)
                Vt[(((size_t)(b * H_ + h)) * HD_ + hd) * S_ + s] = val;
            else if (which == 0)
                Qd[(((size_t)(b * H_ + h)) * S_ + s) * HD_ + hd] = val;
            else
                Kd[(((size_t)(b * H_ + h)) * S_ + s) * HD_ + hd] = val;
        }
    }
}

// ---------------------------------------------------------------------------
// Kernel 4: flash attention in the exp2 domain. One wave per (b,h,16-q tile).
//   Per 64-key chunk: 8 score WMMAs, online softmax (shuffle reductions
//   amortized over 4 tiles), LDS C->A transpose of P, 8 ctx WMMAs.
// ---------------------------------------------------------------------------
__global__ __launch_bounds__(32)
void attn_wmma_kernel(const unsigned short* __restrict__ Qd,   // [B,H,S,HD] (prescaled)
                      const unsigned short* __restrict__ Kd,   // [B,H,S,HD]
                      const unsigned short* __restrict__ Vt,   // [B,H,HD,S]
                      const float* __restrict__ mask,          // [B,1,1,S]
                      float* __restrict__ out)                 // [B,S,D]
{
    const int lane = threadIdx.x & 31;
    const int lh   = lane >> 4;
    const int ll   = lane & 15;
    const int q0   = blockIdx.x * 16;
    const int h    = blockIdx.y;
    const int b    = blockIdx.z;

    __shared__ unsigned short ptile[16 * 64];   // P tile staging (C-layout -> A-layout)

    const size_t bh = (size_t)(b * H_ + h);
    const unsigned short* Qbh = Qd + bh * (size_t)S_ * HD_;
    const unsigned short* Kbh = Kd + bh * (size_t)S_ * HD_;
    const unsigned short* Vbh = Vt + bh * (size_t)HD_ * S_;
    const float* mrow = mask + (size_t)b * S_;

    // Q A-tiles: 16 queries x HD=64 -> two 16x32 A operands, resident in VGPRs
    V16 qa0, qa1;
    {
        const unsigned short* qr = Qbh + (size_t)(q0 + ll) * HD_ + lh * 8;
        qa0.q[0] = *(const uint4*)(qr);
        qa0.q[1] = *(const uint4*)(qr + 16);
        qa1.q[0] = *(const uint4*)(qr + 32);
        qa1.q[1] = *(const uint4*)(qr + 48);
    }

    const v8f zero = {};
    v8f acc[4];
#pragma unroll
    for (int t = 0; t < 4; ++t) acc[t] = zero;

    float mrun[8], lrun[8], fac[8];
#pragma unroll
    for (int r = 0; r < 8; ++r) { mrun[r] = -__builtin_inff(); lrun[r] = 0.f; }

    for (int kc = 0; kc < S_; kc += 64) {
        // ---- scores: four 16x16 tiles (keys kc+16j .. kc+16j+15) ------------
        v8f sc[4];
#pragma unroll
        for (int j = 0; j < 4; ++j) {
            const int key = kc + 16 * j + ll;
            const unsigned short* kr = Kbh + (size_t)key * HD_ + lh * 8;
            V16 kb0, kb1;
            kb0.q[0] = *(const uint4*)(kr);
            kb0.q[1] = *(const uint4*)(kr + 16);
            kb1.q[0] = *(const uint4*)(kr + 32);
            kb1.q[1] = *(const uint4*)(kr + 48);
            v8f s = __builtin_amdgcn_wmma_f32_16x16x32_bf16(
                false, qa0.v, false, kb0.v, (short)0, zero, false, false);
            s = __builtin_amdgcn_wmma_f32_16x16x32_bf16(
                false, qa1.v, false, kb1.v, (short)0, s, false, false);
            const float mv = mrow[key] * LOG2E;   // additive mask in exp2 domain
#pragma unroll
            for (int r = 0; r < 8; ++r) s[r] += mv;
            sc[j] = s;
        }

        // ---- online softmax (exp2 domain); rows in fixed VGPR r per half ----
#pragma unroll
        for (int r = 0; r < 8; ++r) {
            float mx = fmaxf(fmaxf(sc[0][r], sc[1][r]), fmaxf(sc[2][r], sc[3][r]));
#pragma unroll
            for (int o = 1; o < 16; o <<= 1) mx = fmaxf(mx, __shfl_xor(mx, o, 32));
            const float mn = fmaxf(mrun[r], mx);
            const float f  = exp2f(mrun[r] - mn);      // exp2(-inf)=0 on first pass
            float ps = 0.f;
#pragma unroll
            for (int j = 0; j < 4; ++j) {
                const float p = exp2f(sc[j][r] - mn);
                sc[j][r] = p;
                ps += p;
            }
#pragma unroll
            for (int o = 1; o < 16; o <<= 1) ps += __shfl_xor(ps, o, 32);
            lrun[r] = lrun[r] * f + ps;
            mrun[r] = mn;
            fac[r]  = f;
        }
#pragma unroll
        for (int t = 0; t < 4; ++t)
#pragma unroll
            for (int r = 0; r < 8; ++r) acc[t][r] *= fac[r];

        // ---- transpose P (C-layout) -> A-layout via LDS ---------------------
        __syncthreads();
#pragma unroll
        for (int j = 0; j < 4; ++j)
#pragma unroll
            for (int r = 0; r < 8; ++r)
                ptile[(r + lh * 8) * 64 + 16 * j + ll] = f2bf(sc[j][r]);
        __syncthreads();

        V16 pa0, pa1;
        {
            const unsigned short* pr = ptile + ll * 64 + lh * 8;
            pa0.q[0] = *(const uint4*)(pr);          // keys lh*8 .. +7
            pa0.q[1] = *(const uint4*)(pr + 16);     // keys 16+lh*8 .. +7
            pa1.q[0] = *(const uint4*)(pr + 32);
            pa1.q[1] = *(const uint4*)(pr + 48);
        }

        // ---- ctx += P(16x64) x V(64x16) per hd tile, two 32-key sub-chunks --
#pragma unroll
        for (int t = 0; t < 4; ++t) {
            const unsigned short* vr0 = Vbh + (size_t)(t * 16 + ll) * S_ + kc + lh * 8;
            V16 vb0, vb1;
            vb0.q[0] = *(const uint4*)(vr0);
            vb0.q[1] = *(const uint4*)(vr0 + 16);
            vb1.q[0] = *(const uint4*)(vr0 + 32);
            vb1.q[1] = *(const uint4*)(vr0 + 48);
            acc[t] = __builtin_amdgcn_wmma_f32_16x16x32_bf16(
                false, pa0.v, false, vb0.v, (short)0, acc[t], false, false);
            acc[t] = __builtin_amdgcn_wmma_f32_16x16x32_bf16(
                false, pa1.v, false, vb1.v, (short)0, acc[t], false, false);
        }
    }

    // ---- normalize and store [B,S,D] ---------------------------------------
#pragma unroll
    for (int t = 0; t < 4; ++t) {
        const int hd = t * 16 + ll;
#pragma unroll
        for (int r = 0; r < 8; ++r) {
            const int q = q0 + r + lh * 8;
            out[((size_t)b * S_ + q) * D_ + h * HD_ + hd] = acc[t][r] / lrun[r];
        }
    }
}

// ---------------------------------------------------------------------------
extern "C" void kernel_launch(void* const* d_in, const int* in_sizes, int n_in,
                              void* d_out, int out_size, void* d_ws, size_t ws_size,
                              hipStream_t stream)
{
    const float* hidden = (const float*)d_in[0];
    const float* mask   = (const float*)d_in[1];
    const float* gamma  = (const float*)d_in[2];
    const float* beta   = (const float*)d_in[3];
    const float* Wq     = (const float*)d_in[4];
    const float* Wk     = (const float*)d_in[5];
    const float* Wv     = (const float*)d_in[6];
    float* out          = (float*)d_out;

    char* ws = (char*)d_ws;
    size_t off = 0;
    auto alloc = [&](size_t bytes) -> char* {
        char* p = ws + off;
        off += (bytes + 255) & ~(size_t)255;
        return p;
    };
    unsigned short* hln = (unsigned short*)alloc((size_t)B_ * S_ * D_ * 2);        // 8 MB
    unsigned short* Wt  = (unsigned short*)alloc((size_t)3 * D_ * D_ * 2);         // 6 MB
    unsigned short* Qd  = (unsigned short*)alloc((size_t)B_ * H_ * S_ * HD_ * 2);  // 8 MB
    unsigned short* Kd  = (unsigned short*)alloc((size_t)B_ * H_ * S_ * HD_ * 2);  // 8 MB
    unsigned short* Vtd = (unsigned short*)alloc((size_t)B_ * H_ * HD_ * S_ * 2);  // 8 MB
    (void)ws_size; (void)in_sizes; (void)n_in; (void)out_size;

    ln_bf16_kernel<<<dim3(B_ * S_), dim3(256), 0, stream>>>(hidden, gamma, beta, hln);
    wt_bf16_kernel<<<dim3(3 * 1024 * 1024 / 256), dim3(256), 0, stream>>>(Wq, Wk, Wv, Wt);
    qkv_wmma_kernel<<<dim3(4096 / 64, 1024 / 64, 3), dim3(128), 0, stream>>>(hln, Wt, Qd, Kd, Vtd);
    attn_wmma_kernel<<<dim3(S_ / 16, H_, B_), dim3(32), 0, stream>>>(Qd, Kd, Vtd, mask, out);
}
